// HQQSVDLinear_8074538517178
// MI455X (gfx1250) — 8-bit matrix operand order repaired and checked by reading, compiled, not run on hardware
//
#include <hip/hip_runtime.h>
#include <cstdint>

typedef int v8i __attribute__((ext_vector_type(8)));

#define N_TOK 4096
#define IDIM  4096
#define ODIM  4096
#define RANK  32
#define NGRP  64

__device__ __forceinline__ int clamp8i(int v) {
  return v < -128 ? -128 : (v > 127 ? 127 : v);
}

__device__ __forceinline__ uint32_t pack4q(float a, float b, float c, float d, float inv) {
  int q0 = clamp8i((int)rintf(a * inv));
  int q1 = clamp8i((int)rintf(b * inv));
  int q2 = clamp8i((int)rintf(c * inv));
  int q3 = clamp8i((int)rintf(d * inv));
  return (uint32_t)(q0 & 255) | ((uint32_t)(q1 & 255) << 8) |
         ((uint32_t)(q2 & 255) << 16) | ((uint32_t)(q3 & 255) << 24);
}

__device__ __forceinline__ void async_copy_b128(uint32_t lds_addr, const void* gaddr) {
  asm volatile("global_load_async_to_lds_b128 %0, %1, off"
               :: "v"(lds_addr), "v"(gaddr)
               : "memory");
}
__device__ __forceinline__ void wait_async0() {
  asm volatile("s_wait_asynccnt 0x0" ::: "memory");
}

__global__ __launch_bounds__(256)
void quant_x_kernel(const float* __restrict__ x, int8_t* __restrict__ xq,
                    float* __restrict__ scale_x) {
  const int row = blockIdx.x;
  const int tid = threadIdx.x;
  const float4* xr = (const float4*)(x + (size_t)row * IDIM);
  float4 v[4];
  float amax = 0.f;
#pragma unroll
  for (int j = 0; j < 4; ++j) {
    v[j] = xr[j * 256 + tid];
    amax = fmaxf(amax, fmaxf(fmaxf(fabsf(v[j].x), fabsf(v[j].y)),
                             fmaxf(fabsf(v[j].z), fabsf(v[j].w))));
  }
  __shared__ float red[256];
  red[tid] = amax;
  __syncthreads();
#pragma unroll
  for (int s = 128; s > 0; s >>= 1) {
    if (tid < s) red[tid] = fmaxf(red[tid], red[tid + s]);
    __syncthreads();
  }
  const float sc = fmaxf(red[0], 1e-20f) * (1.f / 127.f);
  if (tid == 0) scale_x[row] = sc;
  const float inv = 1.f / sc;
  uint32_t* oq = (uint32_t*)(xq + (size_t)row * IDIM);
#pragma unroll
  for (int j = 0; j < 4; ++j)
    oq[j * 256 + tid] = pack4q(v[j].x, v[j].y, v[j].z, v[j].w, inv);
}

__global__ __launch_bounds__(256)
void dequant_w_kernel(const int* __restrict__ w_packed,
                      const float* __restrict__ svd_up,
                      const float* __restrict__ svd_down,
                      const float* __restrict__ scale,
                      const float* __restrict__ zero_point,
                      int8_t* __restrict__ wq,
                      float* __restrict__ scale_w) {
  const int o = blockIdx.x;
  const int tid = threadIdx.x;
  __shared__ float su[RANK];
  if (tid < RANK) su[tid] = svd_up[o * RANK + tid];
  __syncthreads();

  const int4* pb = (const int4*)(w_packed + (size_t)o * (IDIM / 2));
  int4 b0 = pb[tid * 2 + 0];
  int4 b1 = pb[tid * 2 + 1];
  int bytes[8] = {b0.x, b0.y, b0.z, b0.w, b1.x, b1.y, b1.z, b1.w};

  const float sc = scale[o * NGRP + (tid >> 2)];
  const float zp = zero_point[o * NGRP + (tid >> 2)];
  float w[16];
#pragma unroll
  for (int e = 0; e < 16; ++e) {
    int q = (bytes[e >> 1] >> ((e & 1) * 4)) & 0xF;
    w[e] = ((float)q - zp) * sc;
  }
  const float4* dp = (const float4*)(svd_down) + tid * 4;
#pragma unroll 4
  for (int r = 0; r < RANK; ++r) {
    const float s = su[r];
#pragma unroll
    for (int j = 0; j < 4; ++j) {
      float4 d = dp[(size_t)r * (IDIM / 4) + j];
      w[4 * j + 0] = fmaf(s, d.x, w[4 * j + 0]);
      w[4 * j + 1] = fmaf(s, d.y, w[4 * j + 1]);
      w[4 * j + 2] = fmaf(s, d.z, w[4 * j + 2]);
      w[4 * j + 3] = fmaf(s, d.w, w[4 * j + 3]);
    }
  }
  float amax = 0.f;
#pragma unroll
  for (int e = 0; e < 16; ++e) amax = fmaxf(amax, fabsf(w[e]));
  __shared__ float red[256];
  red[tid] = amax;
  __syncthreads();
#pragma unroll
  for (int s = 128; s > 0; s >>= 1) {
    if (tid < s) red[tid] = fmaxf(red[tid], red[tid + s]);
    __syncthreads();
  }
  const float sw = fmaxf(red[0], 1e-20f) * (1.f / 127.f);
  if (tid == 0) scale_w[o] = sw;
  const float inv = 1.f / sw;
  uint4 pk;
  pk.x = pack4q(w[0], w[1], w[2], w[3], inv);
  pk.y = pack4q(w[4], w[5], w[6], w[7], inv);
  pk.z = pack4q(w[8], w[9], w[10], w[11], inv);
  pk.w = pack4q(w[12], w[13], w[14], w[15], inv);
  *(uint4*)(wq + (size_t)o * IDIM + tid * 16) = pk;
}

#define BM 128
#define BO 128
#define BK 64
#define KSTEPS (IDIM / BK)
#define LSTRIDE 80

__global__ __launch_bounds__(256)
void gemm_i8_wmma_kernel(const int8_t* __restrict__ xq,
                         const int8_t* __restrict__ wq,
                         const float* __restrict__ scale_x,
                         const float* __restrict__ scale_w,
                         const float* __restrict__ bias,
                         float* __restrict__ out) {
  __shared__ __align__(16) uint8_t lds_a[2][BM * LSTRIDE];
  __shared__ __align__(16) uint8_t lds_b[2][BO * LSTRIDE];

  const int tid   = threadIdx.x;
  const int lane  = tid & 31;
  const int wave  = tid >> 5;
  const int waveM = wave >> 2;
  const int waveO = wave & 3;
  const int m0 = blockIdx.y * BM;
  const int o0 = blockIdx.x * BO;

  const int row0 = tid >> 2;
  const int row1 = row0 + 64;
  const int seg  = (tid & 3) * 16;
  const uint8_t* ga0 = (const uint8_t*)xq + (size_t)(m0 + row0) * IDIM + seg;
  const uint8_t* ga1 = (const uint8_t*)xq + (size_t)(m0 + row1) * IDIM + seg;
  const uint8_t* gb0 = (const uint8_t*)wq + (size_t)(o0 + row0) * IDIM + seg;
  const uint8_t* gb1 = (const uint8_t*)wq + (size_t)(o0 + row1) * IDIM + seg;
  const int lofs0 = row0 * LSTRIDE + seg;
  const int lofs1 = row1 * LSTRIDE + seg;

  const uint32_t la_base = (uint32_t)(uintptr_t)&lds_a[0][0];
  const uint32_t lb_base = (uint32_t)(uintptr_t)&lds_b[0][0];

  v8i acc[4][2];
#pragma unroll
  for (int i = 0; i < 4; ++i)
#pragma unroll
    for (int j = 0; j < 2; ++j) {
      v8i z = {0, 0, 0, 0, 0, 0, 0, 0};
      acc[i][j] = z;
    }

  async_copy_b128(la_base + lofs0, ga0);
  async_copy_b128(la_base + lofs1, ga1);
  async_copy_b128(lb_base + lofs0, gb0);
  async_copy_b128(lb_base + lofs1, gb1);
  wait_async0();
  __syncthreads();

  const int arowbase = (waveM * 64 + (lane & 15)) * LSTRIDE + ((lane >> 4) << 4);
  const int browbase = (waveO * 32 + (lane & 15)) * LSTRIDE + ((lane >> 4) << 4);

  for (int ks = 0; ks < KSTEPS; ++ks) {
    const int buf = ks & 1;
    if (ks + 1 < KSTEPS) {
      const int koff = (ks + 1) * BK;
      const uint32_t ab = la_base + (buf ^ 1) * (BM * LSTRIDE);
      const uint32_t bb = lb_base + (buf ^ 1) * (BO * LSTRIDE);
      async_copy_b128(ab + lofs0, ga0 + koff);
      async_copy_b128(ab + lofs1, ga1 + koff);
      async_copy_b128(bb + lofs0, gb0 + koff);
      async_copy_b128(bb + lofs1, gb1 + koff);
    }

    v8i af[4], bf[2];
#pragma unroll
    for (int mf = 0; mf < 4; ++mf) {
      const uint8_t* p = &lds_a[buf][arowbase + mf * 16 * LSTRIDE];
      uint64_t d0 = *(const uint64_t*)(p);
      uint64_t d1 = *(const uint64_t*)(p + 8);
      uint64_t d2 = *(const uint64_t*)(p + 32);
      uint64_t d3 = *(const uint64_t*)(p + 40);
      af[mf][0] = (int)(uint32_t)d0; af[mf][1] = (int)(uint32_t)(d0 >> 32);
      af[mf][2] = (int)(uint32_t)d1; af[mf][3] = (int)(uint32_t)(d1 >> 32);
      af[mf][4] = (int)(uint32_t)d2; af[mf][5] = (int)(uint32_t)(d2 >> 32);
      af[mf][6] = (int)(uint32_t)d3; af[mf][7] = (int)(uint32_t)(d3 >> 32);
    }
#pragma unroll
    for (int of = 0; of < 2; ++of) {
      const uint8_t* p = &lds_b[buf][browbase + of * 16 * LSTRIDE];
      uint4 lo = *(const uint4*)(p);
      uint4 hi = *(const uint4*)(p + 32);
      bf[of][0] = (int)lo.x; bf[of][1] = (int)lo.y;
      bf[of][2] = (int)lo.z; bf[of][3] = (int)lo.w;
      bf[of][4] = (int)hi.x; bf[of][5] = (int)hi.y;
      bf[of][6] = (int)hi.z; bf[of][7] = (int)hi.w;
    }

#pragma unroll
    for (int mf = 0; mf < 4; ++mf)
#pragma unroll
      for (int of = 0; of < 2; ++of)
        acc[mf][of] = __builtin_amdgcn_wmma_i32_16x16x64_iu8(
            true, af[mf], true, bf[of], acc[mf][of], false, false);

    wait_async0();
    __syncthreads();
  }

  const int ocol0 = o0 + waveO * 32 + (lane & 15);
  const int mbase = m0 + waveM * 64 + ((lane >> 4) << 3);
  const float sw0 = scale_w[ocol0];
  const float sw1 = scale_w[ocol0 + 16];
  const float bs0 = bias[ocol0];
  const float bs1 = bias[ocol0 + 16];
#pragma unroll
  for (int mf = 0; mf < 4; ++mf) {
#pragma unroll
    for (int v = 0; v < 8; ++v) {
      const int n = mbase + mf * 16 + v;
      const float sx = scale_x[n];
      float* orow = out + (size_t)n * ODIM;
      orow[ocol0]      = (float)acc[mf][0][v] * sx * sw0 + bs0;
      orow[ocol0 + 16] = (float)acc[mf][1][v] * sx * sw1 + bs1;
    }
  }
}

extern "C" void kernel_launch(void* const* d_in, const int* in_sizes, int n_in,
                              void* d_out, int out_size, void* d_ws, size_t ws_size,
                              hipStream_t stream) {
  (void)in_sizes; (void)n_in; (void)out_size; (void)ws_size;
  const float* x        = (const float*)d_in[0];
  const int*   w_packed = (const int*)d_in[1];
  const float* svd_up   = (const float*)d_in[2];
  const float* svd_down = (const float*)d_in[3];
  const float* scale    = (const float*)d_in[4];
  const float* zp       = (const float*)d_in[5];
  const float* bias     = (const float*)d_in[6];
  float* out = (float*)d_out;

  uint8_t* ws = (uint8_t*)d_ws;
  int8_t* xq = (int8_t*)ws;
  int8_t* wq = (int8_t*)(ws + (size_t)N_TOK * IDIM);
  float* scale_x = (float*)(ws + (size_t)N_TOK * IDIM + (size_t)ODIM * IDIM);
  float* scale_w = scale_x + N_TOK;

  quant_x_kernel<<<N_TOK, 256, 0, stream>>>(x, xq, scale_x);
  dequant_w_kernel<<<ODIM, 256, 0, stream>>>(w_packed, svd_up, svd_down,
                                             scale, zp, wq, scale_w);
  gemm_i8_wmma_kernel<<<dim3(ODIM / BO, N_TOK / BM), 256, 0, stream>>>(
      xq, wq, scale_x, scale_w, bias, out);
}
